// LStoM_60945585930619
// MI455X (gfx1250) — matrix operand order repaired and checked by reading, not compiled, not run on hardware
//
#include <hip/hip_runtime.h>
#include <hip/hip_bf16.h>
#include <stdint.h>

// ---------------------------------------------------------------------------
// CDNA5 (gfx1250) bf16 WMMA LSTM.
//   z = x_t @ wih.T + h @ whh.T + b   -> i,f,g,o gates -> c,h update.
//   8 workgroups per layer: (4 batch blocks of M=16) x (2 directions).
//   whh B-fragments resident in VGPRs; wih staged in LDS; h double-buffered
//   in LDS (bf16, padded stride for bank-conflict-free ds_load_b128).
// ---------------------------------------------------------------------------

typedef __attribute__((ext_vector_type(16))) __bf16 v16bf;
typedef __attribute__((ext_vector_type(8)))  float  v8f;

#define S_LEN 4096
#define BATCH 64
#define HID   128
#define HSTR  136   // padded LDS row stride (bf16 elems): 272B = 17*16B -> 64-bank friendly

static __device__ __forceinline__ __bf16 f2bf(float f) {
  union { float f; uint32_t u; } in; in.f = f;
  uint32_t u = in.u;
  uint32_t r = (u + 0x7fffu + ((u >> 16) & 1u)) >> 16;   // round-to-nearest-even
  union { uint16_t s; __bf16 b; } out; out.s = (uint16_t)r;
  return out.b;
}
static __device__ __forceinline__ float bf2f(__bf16 b) {
  union { __bf16 b; uint16_t s; } in; in.b = b;
  union { uint32_t u; float f; } out; out.u = ((uint32_t)in.s) << 16;
  return out.f;
}

// A-fragment (16x32 bf16): lane m=lane&15 holds row m; per lane 16 elems are
// K = {koff..koff+7, koff+16..koff+23} with koff = (lane>=16)*8.
// -> two 16B chunks at +0 and +16 elements from the per-lane base.
static __device__ __forceinline__ v16bf load_a_frag(const __bf16* p) {
  union { uint4 q[2]; v16bf v; } u;
  u.q[0] = *(const uint4*)(p);
  u.q[1] = *(const uint4*)(p + 16);
  return u.v;
}
// B-fragment (32x16 bf16): lane L holds column N=L&15 of the tile; 16 K-values
// per lane, same K order as A (pre-packed) -> two contiguous 16B chunks.
static __device__ __forceinline__ v16bf load_b_frag(const __bf16* p) {
  union { uint4 q[2]; v16bf v; } u;
  u.q[0] = *(const uint4*)(p);
  u.q[1] = *(const uint4*)(p + 8);
  return u.v;
}

static __device__ __forceinline__ float fast_sigmoid(float x) {
  return 1.f / (1.f + __expf(-x));
}
static __device__ __forceinline__ float fast_tanh(float x) {
  float ax = fabsf(x);
  float e2 = __expf(-2.f * ax);          // in (0,1], never overflows
  float th = (1.f - e2) / (1.f + e2);
  return copysignf(th, x);
}

// ---------------------------------------------------------------------------
// Pack W [N=512, K] (row-major f32, computing x@W.T) into bf16 B-fragment
// order: dst[tile*512 + lane*16 + j] = W[nt*16 + (lane&15)][kt*32 + 8*(lane>>4) + (j&7) + 16*(j>>3)],
// tile = nt*(K/32) + kt.
// ---------------------------------------------------------------------------
__global__ void pack_weight_kernel(const float* __restrict__ src,
                                   __bf16* __restrict__ dst, int K, int total) {
  int ktn = K >> 5;
  for (int i = blockIdx.x * blockDim.x + threadIdx.x; i < total;
       i += gridDim.x * blockDim.x) {
    int t    = i >> 9;
    int lane = (i >> 4) & 31;
    int j    = i & 15;
    int nt   = t / ktn;
    int kt   = t - nt * ktn;
    dst[i] = f2bf(src[(nt * 16 + (lane & 15)) * K + kt * 32 + (j & 7) + ((lane >> 4) << 3) + ((j >> 3) << 4)]);
  }
}

__global__ void cast_bf16_kernel(const float* __restrict__ src,
                                 __bf16* __restrict__ dst, int n) {
  for (int i = blockIdx.x * blockDim.x + threadIdx.x; i < n;
       i += gridDim.x * blockDim.x)
    dst[i] = f2bf(src[i]);
}

// ---------------------------------------------------------------------------
// One full LSTM layer (both pseudo-directions stepped forward in time).
// grid = 8 blocks: blockIdx.x = (batch_block<<1) | dir. block = 256 (8 waves).
// Wave w owns h columns [16w, 16w+16); its four gate z-tiles are N-tiles
// {w, w+8, w+16, w+24}. Input xin: [S][64][DIN] bf16 row-major.
// Output ys: [S][64][256] bf16; dir d writes columns [128d, 128d+128).
// ---------------------------------------------------------------------------
template <int DIN>
__global__ void __launch_bounds__(256, 1)
lstm_layer_kernel(const __bf16* __restrict__ xin,
                  const __bf16* __restrict__ wih_pf, const __bf16* __restrict__ wih_pb,
                  const __bf16* __restrict__ whh_pf, const __bf16* __restrict__ whh_pb,
                  const float* __restrict__ bias_f, const float* __restrict__ bias_b,
                  __bf16* __restrict__ ys) {
  constexpr int KT_IN = DIN / 32;
  const int dir = blockIdx.x & 1;
  const int mb  = blockIdx.x >> 1;
  const int tid  = threadIdx.x;
  const int w    = tid >> 5;
  const int lane = tid & 31;
  const int m    = lane & 15;
  const int koff = (lane >> 4) * 8;
  const int rowoff = (lane >> 4) << 3;  // D-frag: elem e -> row e (+8 for hi lanes)

  const __bf16* wihp = dir ? wih_pb : wih_pf;
  const __bf16* whhp = dir ? whh_pb : whh_pf;
  const float*  bias = dir ? bias_b : bias_f;

  extern __shared__ __align__(16) char smem[];
  __bf16* wihL = (__bf16*)smem;              // 512*DIN bf16, B-frag order
  __bf16* hbuf = wihL + 512 * DIN;           // 2 x [16][HSTR] bf16

  // Stage wih into LDS; zero both h buffers.
  {
    const uint4* src = (const uint4*)wihp;
    uint4* dst = (uint4*)wihL;
    const int nv = 512 * DIN / 8;
    for (int i = tid; i < nv; i += 256) dst[i] = src[i];
    for (int i = tid; i < 2 * 16 * HSTR; i += 256) hbuf[i] = f2bf(0.f);
  }

  // Per-gate bias for this lane's output column.
  float bval[4];
#pragma unroll
  for (int g = 0; g < 4; ++g) bval[g] = bias[g * HID + w * 16 + m];

  // whh B-fragments: resident in VGPRs for all 4096 steps (16 x 8 = 128 VGPRs).
  v16bf wfr[16];
#pragma unroll
  for (int g = 0; g < 4; ++g)
#pragma unroll
    for (int kt = 0; kt < 4; ++kt)
      wfr[g * 4 + kt] =
          load_b_frag(whhp + (size_t)((g * 8 + w) * 4 + kt) * 512 + lane * 16);

  float cst[8];
#pragma unroll
  for (int e = 0; e < 8; ++e) cst[e] = 0.f;

  __syncthreads();

  for (int t = 0; t < S_LEN; ++t) {
    const __bf16* hcur = hbuf + (t & 1) * (16 * HSTR);
    __bf16*       hnx  = hbuf + ((t + 1) & 1) * (16 * HSTR);

    union { v8f v; float f[8]; } acc[4];
#pragma unroll
    for (int g = 0; g < 4; ++g)
#pragma unroll
      for (int e = 0; e < 8; ++e) acc[g].f[e] = bval[g];

    // ---- input contribution: z += x_t @ wih.T ----
    const __bf16* xrow =
        xin + ((size_t)t * BATCH + mb * 16 + m) * DIN + koff;
    if (t + 1 < S_LEN) __builtin_prefetch(xrow + BATCH * DIN, 0, 3);
#pragma unroll
    for (int kt = 0; kt < KT_IN; ++kt) {
      v16bf a = load_a_frag(xrow + kt * 32);
#pragma unroll
      for (int g = 0; g < 4; ++g) {
        v16bf bf = load_b_frag(wihL + (size_t)((g * 8 + w) * KT_IN + kt) * 512 +
                               lane * 16);
        acc[g].v = __builtin_amdgcn_wmma_f32_16x16x32_bf16(
            false, a, false, bf, (short)0, acc[g].v, false, false);
      }
    }

    // ---- recurrent contribution: z += h_{t-1} @ whh.T ----
#pragma unroll
    for (int kt = 0; kt < 4; ++kt) {
      v16bf a = load_a_frag(hcur + m * HSTR + kt * 32 + koff);
#pragma unroll
      for (int g = 0; g < 4; ++g)
        acc[g].v = __builtin_amdgcn_wmma_f32_16x16x32_bf16(
            false, a, false, wfr[g * 4 + kt], (short)0, acc[g].v, false, false);
    }

    // ---- gates + state update (fp32 VALU on D-fragments) ----
    size_t ybase = ((size_t)t * BATCH + mb * 16) * 256 + (size_t)dir * 128 +
                   (size_t)(w * 16 + m);
#pragma unroll
    for (int e = 0; e < 8; ++e) {
      float iv = fast_sigmoid(acc[0].f[e]);
      float fv = fast_sigmoid(acc[1].f[e]);
      float gv = fast_tanh(acc[2].f[e]);
      float ov = fast_sigmoid(acc[3].f[e]);
      float c  = fv * cst[e] + iv * gv;
      cst[e]   = c;
      float hv = ov * fast_tanh(c);
      __bf16 hb = f2bf(hv);
      int row = e + rowoff;
      hnx[row * HSTR + w * 16 + m] = hb;            // next-step A operand
      ys[ybase + (size_t)row * 256] = hb;           // layer output
    }
    __syncthreads();
  }
}

// ---------------------------------------------------------------------------
// out[s,b] = sigmoid( ys1[s,b,:] . fc_w + fc_b )
// ---------------------------------------------------------------------------
__global__ void fc_kernel(const __bf16* __restrict__ ys,
                          const float* __restrict__ w,
                          const float* __restrict__ b,
                          float* __restrict__ out, int n) {
  int i = blockIdx.x * blockDim.x + threadIdx.x;
  if (i >= n) return;
  const __bf16* r = ys + (size_t)i * 256;
  float s = b[0];
#pragma unroll 8
  for (int c = 0; c < 256; ++c) s += bf2f(r[c]) * w[c];
  out[i] = 1.f / (1.f + __expf(-s));
}

// ---------------------------------------------------------------------------
extern "C" void kernel_launch(void* const* d_in, const int* in_sizes, int n_in,
                              void* d_out, int out_size, void* d_ws,
                              size_t ws_size, hipStream_t stream) {
  (void)in_sizes; (void)n_in; (void)out_size; (void)ws_size;

  const float* x     = (const float*)d_in[0];
  const float* wih0f = (const float*)d_in[1];
  const float* whh0f = (const float*)d_in[2];
  const float* b0f   = (const float*)d_in[3];
  const float* wih0b = (const float*)d_in[4];
  const float* whh0b = (const float*)d_in[5];
  const float* b0b   = (const float*)d_in[6];
  const float* wih1f = (const float*)d_in[7];
  const float* whh1f = (const float*)d_in[8];
  const float* b1f   = (const float*)d_in[9];
  const float* wih1b = (const float*)d_in[10];
  const float* whh1b = (const float*)d_in[11];
  const float* b1b   = (const float*)d_in[12];
  const float* fcw   = (const float*)d_in[13];
  const float* fcb   = (const float*)d_in[14];

  char* ws = (char*)d_ws;
  size_t off = 0;
  auto carve = [&](size_t bytes) -> void* {
    off = (off + 255) & ~(size_t)255;
    void* p = ws + off;
    off += bytes;
    return p;
  };

  __bf16* xbf = (__bf16*)carve((size_t)S_LEN * BATCH * 96 * 2);   // 50 MB
  __bf16* ys0 = (__bf16*)carve((size_t)S_LEN * BATCH * 256 * 2);  // 134 MB
  __bf16* ys1 = (__bf16*)carve((size_t)S_LEN * BATCH * 256 * 2);  // 134 MB
  __bf16* wih0pf = (__bf16*)carve((size_t)512 * 96 * 2);
  __bf16* wih0pb = (__bf16*)carve((size_t)512 * 96 * 2);
  __bf16* whh0pf = (__bf16*)carve((size_t)512 * 128 * 2);
  __bf16* whh0pb = (__bf16*)carve((size_t)512 * 128 * 2);
  __bf16* wih1pf = (__bf16*)carve((size_t)512 * 256 * 2);
  __bf16* wih1pb = (__bf16*)carve((size_t)512 * 256 * 2);
  __bf16* whh1pf = (__bf16*)carve((size_t)512 * 128 * 2);
  __bf16* whh1pb = (__bf16*)carve((size_t)512 * 128 * 2);

  // Phase 0: convert input to bf16; pack all weights into B-fragment order.
  cast_bf16_kernel<<<2048, 256, 0, stream>>>(x, xbf, S_LEN * BATCH * 96);
  pack_weight_kernel<<<192, 256, 0, stream>>>(wih0f, wih0pf, 96, 512 * 96);
  pack_weight_kernel<<<192, 256, 0, stream>>>(wih0b, wih0pb, 96, 512 * 96);
  pack_weight_kernel<<<256, 256, 0, stream>>>(whh0f, whh0pf, 128, 512 * 128);
  pack_weight_kernel<<<256, 256, 0, stream>>>(whh0b, whh0pb, 128, 512 * 128);
  pack_weight_kernel<<<512, 256, 0, stream>>>(wih1f, wih1pf, 256, 512 * 256);
  pack_weight_kernel<<<512, 256, 0, stream>>>(wih1b, wih1pb, 256, 512 * 256);
  pack_weight_kernel<<<256, 256, 0, stream>>>(whh1f, whh1pf, 128, 512 * 128);
  pack_weight_kernel<<<256, 256, 0, stream>>>(whh1b, whh1pb, 128, 512 * 128);

  // Phase 1/2: the two recurrent layers (8 WGs each own a full cell sequence).
  size_t lds0 = (size_t)512 * 96 * 2 + (size_t)2 * 16 * HSTR * 2;    // ~105 KB
  size_t lds1 = (size_t)512 * 256 * 2 + (size_t)2 * 16 * HSTR * 2;   // ~265 KB
  lstm_layer_kernel<96><<<8, 256, lds0, stream>>>(
      xbf, wih0pf, wih0pb, whh0pf, whh0pb, b0f, b0b, ys0);
  lstm_layer_kernel<256><<<8, 256, lds1, stream>>>(
      ys0, wih1pf, wih1pb, whh1pf, whh1pb, b1f, b1b, ys1);

  // Phase 3: FC + sigmoid epilogue.
  fc_kernel<<<(S_LEN * BATCH + 255) / 256, 256, 0, stream>>>(
      ys1, fcw, fcb, (float*)d_out, S_LEN * BATCH);
}
